// LinearAttention_59270548684917
// MI455X (gfx1250) — matrix operand order repaired and checked by reading, not compiled, not run on hardware
//
#include <hip/hip_runtime.h>
#include <math.h>

// ---------------------------------------------------------------------------
// LinearAttention for MI455X (gfx1250): bf16 WMMA pipeline, fp32 softmax/norm.
// ---------------------------------------------------------------------------

typedef __bf16 bf16;
typedef __attribute__((ext_vector_type(8)))  __bf16 bf16x8;
typedef __attribute__((ext_vector_type(16))) __bf16 bf16x16;
typedef __attribute__((ext_vector_type(8)))  float  floatx8;

#define B_   8
#define C_   256
#define N_   16384
#define H_   4
#define DH_  32
#define HID_ 128
#define NT_  64        // columns (n) per workgroup tile

// workspace layout (bytes)
static constexpr size_t QB_OFF   = 0;                                     // bf16 [B][128][N]
static constexpr size_t KB_OFF   = QB_OFF   + (size_t)B_*HID_*N_*2;       // bf16 [B][128][N]
static constexpr size_t VB_OFF   = KB_OFF   + (size_t)B_*HID_*N_*2;       // bf16 [B][128][N]
static constexpr size_t WQKV_OFF = VB_OFF   + (size_t)B_*HID_*N_*2;       // bf16 [384][256]
static constexpr size_t WOUT_OFF = WQKV_OFF + (size_t)384*256*2;          // bf16 [256][128]
static constexpr size_t KMAX_OFF = WOUT_OFF + (size_t)256*128*2;          // f32 [B*128]
static constexpr size_t KSUM_OFF = KMAX_OFF + (size_t)B_*HID_*4;          // f32 [B*128]
static constexpr size_t CTX_OFF  = KSUM_OFF + (size_t)B_*HID_*4;          // f32 [B][H][32][32]

__device__ __forceinline__ floatx8 zero8() {
  floatx8 z = {0.f,0.f,0.f,0.f,0.f,0.f,0.f,0.f};
  return z;
}
__device__ __forceinline__ bf16x16 join8(bf16x8 a, bf16x8 b) {
  return __builtin_shufflevector(a, b, 0,1,2,3,4,5,6,7,8,9,10,11,12,13,14,15);
}
__device__ __forceinline__ floatx8 wmma_bf16(bf16x16 a, bf16x16 b, floatx8 c) {
  // D = A(16x32) * B(32x16) + C, fp32 accum
  return __builtin_amdgcn_wmma_f32_16x16x32_bf16(false, a, false, b, (short)0, c, false, false);
}

// ---------------------------------------------------------------------------
// Kernel 0: weight fp32->bf16 conversion + zero context accumulator
// ---------------------------------------------------------------------------
__global__ __launch_bounds__(256)
void la_prep(const float* __restrict__ wqkv, const float* __restrict__ wout,
             bf16* __restrict__ wqkv_b, bf16* __restrict__ wout_b,
             float* __restrict__ ctxg) {
  int i = blockIdx.x * blockDim.x + threadIdx.x;
  int stride = gridDim.x * blockDim.x;
  for (int j = i; j < 384*256; j += stride) wqkv_b[j] = (bf16)wqkv[j];
  for (int j = i; j < 256*128; j += stride) wout_b[j] = (bf16)wout[j];
  for (int j = i; j < B_*H_*DH_*DH_; j += stride) ctxg[j] = 0.f;
}

// ---------------------------------------------------------------------------
// Kernel 1: qkv[384,NT] = w_qkv(bf16) @ x(tile)  -> store q,k,v bf16
//   per block: one (b, 64-column tile). 8 waves, each 3 M-tiles x 4 N-tiles.
// ---------------------------------------------------------------------------
__global__ __launch_bounds__(256)
void la_qkv(const float* __restrict__ x, const bf16* __restrict__ wq,
            bf16* __restrict__ qb, bf16* __restrict__ kb, bf16* __restrict__ vb) {
  __shared__ __align__(32) bf16 xT[NT_][C_];   // 32 KB, [n][c] so B-frags are K-contiguous

  const int b  = blockIdx.x / (N_ / NT_);
  const int n0 = (blockIdx.x % (N_ / NT_)) * NT_;
  const int tid = threadIdx.x;

  // load + transpose + convert x tile (coalesced over n)
  const float* xb = x + (size_t)b * C_ * N_;
#pragma unroll 4
  for (int it = 0; it < (NT_ * C_) / 256; ++it) {
    int idx = it * 256 + tid;
    int c = idx >> 6;          // /NT_
    int n = idx & (NT_ - 1);
    xT[n][c] = (bf16)xb[(size_t)c * N_ + n0 + n];
  }
  __syncthreads();

  const int lane = tid & 31, wv = tid >> 5;
  const int lr = lane & 15, lh = lane >> 4;

  floatx8 acc[3][4];
#pragma unroll
  for (int r = 0; r < 3; ++r)
#pragma unroll
    for (int nt = 0; nt < 4; ++nt) acc[r][nt] = zero8();

#pragma unroll 2
  for (int kb_ = 0; kb_ < C_ / 32; ++kb_) {
    const int kbase = kb_ * 32;
    bf16x16 Bfr[4];
#pragma unroll
    for (int nt = 0; nt < 4; ++nt)
      Bfr[nt] = *(const bf16x16*)&xT[nt * 16 + lr][kbase + lh * 16];
    bf16x16 Afr[3];
#pragma unroll
    for (int r = 0; r < 3; ++r) {
      int m = (wv + r * 8) * 16 + lr;            // row of w_qkv
      const bf16* p = wq + (size_t)m * C_ + kbase + lh * 16;
      bf16x8 lo = *(const bf16x8*)p;
      bf16x8 hi = *(const bf16x8*)(p + 8);
      Afr[r] = join8(lo, hi);
    }
#pragma unroll
    for (int r = 0; r < 3; ++r)
#pragma unroll
      for (int nt = 0; nt < 4; ++nt)
        acc[r][nt] = wmma_bf16(Afr[r], Bfr[nt], acc[r][nt]);
  }

  bf16* dsts[3] = {qb, kb, vb};                  // rows 0..127=q, 128..255=k, 256..383=v
#pragma unroll
  for (int r = 0; r < 3; ++r) {
    bf16* dst = dsts[r];
#pragma unroll
    for (int nt = 0; nt < 4; ++nt) {
      int n = n0 + nt * 16 + lr;
#pragma unroll
      for (int gg = 0; gg < 8; ++gg) {
        int m = wv * 16 + gg + 8 * lh;           // local row within q/k/v
        dst[(size_t)(b * HID_ + m) * N_ + n] = (bf16)acc[r][nt][gg];
      }
    }
  }
}

// ---------------------------------------------------------------------------
// Kernel 2: per-row (b*128 rows) max and sum(exp(k - max)) over n=16384
// ---------------------------------------------------------------------------
__global__ __launch_bounds__(256)
void la_kstats(const bf16* __restrict__ kb, float* __restrict__ kmax,
               float* __restrict__ ksum) {
  const int r = blockIdx.x;                      // 0 .. B*128-1
  const bf16* row = kb + (size_t)r * N_;
  const int tid = threadIdx.x;

  float v[64];
#pragma unroll
  for (int i = 0; i < 64; ++i) v[i] = (float)row[i * 256 + tid];

  float m = -1e30f;
#pragma unroll
  for (int i = 0; i < 64; ++i) m = fmaxf(m, v[i]);
#pragma unroll
  for (int off = 16; off > 0; off >>= 1) m = fmaxf(m, __shfl_xor(m, off));

  __shared__ float redm[8];
  __shared__ float reds[8];
  if ((tid & 31) == 0) redm[tid >> 5] = m;
  __syncthreads();
  float rowmax = redm[0];
#pragma unroll
  for (int j = 1; j < 8; ++j) rowmax = fmaxf(rowmax, redm[j]);

  float s = 0.f;
#pragma unroll
  for (int i = 0; i < 64; ++i) s += __expf(v[i] - rowmax);
#pragma unroll
  for (int off = 16; off > 0; off >>= 1) s += __shfl_xor(s, off);
  if ((tid & 31) == 0) reds[tid >> 5] = s;
  __syncthreads();
  if (tid == 0) {
    float tot = 0.f;
#pragma unroll
    for (int j = 0; j < 8; ++j) tot += reds[j];
    kmax[r] = rowmax;
    ksum[r] = tot;
  }
}

// ---------------------------------------------------------------------------
// Kernel 3: ctx[b,h][32,32] += exp(k - kmax)[32 x Kslice] @ v^T[Kslice x 32]
//   grid = (b*h) x 16 K-splits of 1024; wave handles 128-K sub-slice.
// ---------------------------------------------------------------------------
#define KSPLIT_ 16
__global__ __launch_bounds__(256)
void la_ctx(const bf16* __restrict__ kb, const bf16* __restrict__ vb,
            const float* __restrict__ kmax, float* __restrict__ ctxg) {
  const int bh = blockIdx.x >> 4;                // /KSPLIT_
  const int ks = blockIdx.x & (KSPLIT_ - 1);
  const int b = bh >> 2, h = bh & 3;
  const int tid = threadIdx.x, lane = tid & 31, wv = tid >> 5;
  const int lr = lane & 15, lh = lane >> 4;

  const size_t rowbase = (size_t)(b * HID_ + h * DH_);
  float kmx[2];
#pragma unroll
  for (int dt = 0; dt < 2; ++dt)
    kmx[dt] = kmax[b * HID_ + h * DH_ + dt * 16 + lr];

  floatx8 acc[2][2];
#pragma unroll
  for (int dt = 0; dt < 2; ++dt)
#pragma unroll
    for (int et = 0; et < 2; ++et) acc[dt][et] = zero8();

  const int nstart = ks * (N_ / KSPLIT_) + wv * 128;
#pragma unroll 2
  for (int it = 0; it < 4; ++it) {
    const int kpos = nstart + it * 32;
    bf16x16 Afr[2];
#pragma unroll
    for (int dt = 0; dt < 2; ++dt) {
      const bf16* kr = kb + (rowbase + dt * 16 + lr) * N_ + kpos + lh * 16;
      __builtin_prefetch(kr + 32, 0, 1);
      bf16x8 lo = *(const bf16x8*)kr;
      bf16x8 hi = *(const bf16x8*)(kr + 8);
      bf16x16 a;
#pragma unroll
      for (int i = 0; i < 8; ++i) a[i]     = (bf16)__expf((float)lo[i] - kmx[dt]);
#pragma unroll
      for (int i = 0; i < 8; ++i) a[8 + i] = (bf16)__expf((float)hi[i] - kmx[dt]);
      Afr[dt] = a;
    }
    bf16x16 Bfr[2];
#pragma unroll
    for (int et = 0; et < 2; ++et) {
      const bf16* vr = vb + (rowbase + et * 16 + lr) * N_ + kpos + lh * 16;
      __builtin_prefetch(vr + 32, 0, 1);
      Bfr[et] = *(const bf16x16*)vr;
    }
#pragma unroll
    for (int dt = 0; dt < 2; ++dt)
#pragma unroll
      for (int et = 0; et < 2; ++et)
        acc[dt][et] = wmma_bf16(Afr[dt], Bfr[et], acc[dt][et]);
  }

  // cross-wave reduce in LDS, then one atomicAdd per element to global
  __shared__ float sctx[DH_ * DH_];
  for (int i = tid; i < DH_ * DH_; i += 256) sctx[i] = 0.f;
  __syncthreads();
#pragma unroll
  for (int dt = 0; dt < 2; ++dt)
#pragma unroll
    for (int et = 0; et < 2; ++et)
#pragma unroll
      for (int gg = 0; gg < 8; ++gg) {
        int d = dt * 16 + gg + 8 * lh;
        int e = et * 16 + lr;
        atomicAdd(&sctx[d * DH_ + e], acc[dt][et][gg]);
      }
  __syncthreads();
  float* cg = ctxg + (size_t)(b * H_ + h) * DH_ * DH_;
  for (int i = tid; i < DH_ * DH_; i += 256) atomicAdd(&cg[i], sctx[i]);
}

// ---------------------------------------------------------------------------
// Kernel 4: q-softmax -> attn = ctx^T @ qs (block-diag) -> w_out GEMM
//           -> +bias -> RMS-norm over c=256 -> fp32 out
// ---------------------------------------------------------------------------
__global__ __launch_bounds__(256)
void la_out(const bf16* __restrict__ qb, const bf16* __restrict__ woutb,
            const float* __restrict__ ctxg, const float* __restrict__ ksum,
            const float* __restrict__ bias, const float* __restrict__ gvec,
            float* __restrict__ out) {
  __shared__ __align__(32) unsigned char smem[32768 + 16384 + 8192 + 256];
  float* qT   = (float*)smem;                    // [64][128] fp32 (32 KB)
  bf16* attnT = (bf16*)smem;                     // [64][128] bf16, reuses qT region
  bf16* qsT   = (bf16*)(smem + 32768);           // [64][128] bf16 (16 KB)
  bf16* ctxT  = (bf16*)(smem + 32768 + 16384);   // [4][32(e)][32(d)] bf16 (8 KB)
  float* ssq  = (float*)(smem + 32768 + 16384 + 8192); // [64]

  const int b  = blockIdx.x / (N_ / NT_);
  const int n0 = (blockIdx.x % (N_ / NT_)) * NT_;
  const int tid = threadIdx.x;

  if (tid < NT_) ssq[tid] = 0.f;

  // stage q tile [128][64] -> qT[n][c] fp32
  const bf16* qbb = qb + (size_t)b * HID_ * N_;
#pragma unroll 4
  for (int it = 0; it < (NT_ * HID_) / 256; ++it) {
    int idx = it * 256 + tid;
    int c = idx >> 6;
    int n = idx & (NT_ - 1);
    qT[n * HID_ + c] = (float)qbb[(size_t)c * N_ + n0 + n];
  }
  // stage ctx^T / ksum -> bf16
  for (int i = tid; i < H_ * DH_ * DH_; i += 256) {
    int h = i >> 10, rem = i & 1023, e = rem >> 5, d = rem & 31;
    float val = ctxg[((size_t)(b * H_ + h) * DH_ + d) * DH_ + e] /
                ksum[b * HID_ + h * DH_ + d];
    ctxT[(h * DH_ + e) * DH_ + d] = (bf16)val;
  }
  __syncthreads();

  // per-column softmax over each 32-wide head group, * 1/sqrt(32)
  {
    int n = tid >> 2, h = tid & 3;
    float* p = qT + n * HID_ + h * DH_;
    float mx = -1e30f;
#pragma unroll
    for (int i = 0; i < DH_; ++i) mx = fmaxf(mx, p[i]);
    float s = 0.f;
#pragma unroll
    for (int i = 0; i < DH_; ++i) { float e = __expf(p[i] - mx); p[i] = e; s += e; }
    float inv = 0.17677669529663688f / s;        // scale / sum
#pragma unroll
    for (int i = 0; i < DH_; ++i) qsT[n * HID_ + h * DH_ + i] = (bf16)(p[i] * inv);
  }
  __syncthreads();

  const int lane = tid & 31, wv = tid >> 5;
  const int lr = lane & 15, lh = lane >> 4;

  // GEMM1: attn[m=wv*16.., n] = ctxT_h @ qs  (K = 32, one WMMA per tile)
  {
    const int h = wv >> 1;
    const int ebase = (wv & 1) * 16;
    bf16x8 lo = *(const bf16x8*)&ctxT[(h * DH_ + ebase + lr) * DH_ + lh * 16];
    bf16x8 hi = *(const bf16x8*)&ctxT[(h * DH_ + ebase + lr) * DH_ + 8 + lh * 16];
    bf16x16 Afr = join8(lo, hi);
    floatx8 acc1[4];
#pragma unroll
    for (int nt = 0; nt < 4; ++nt) {
      bf16x16 Bfr = *(const bf16x16*)&qsT[(nt * 16 + lr) * HID_ + h * DH_ + lh * 16];
      acc1[nt] = wmma_bf16(Afr, Bfr, zero8());
    }
    // write attnT[n][m] bf16 (overwrites qT region; qT no longer needed)
#pragma unroll
    for (int nt = 0; nt < 4; ++nt) {
      int n = nt * 16 + lr;
#pragma unroll
      for (int gg = 0; gg < 8; ++gg) {
        int m = wv * 16 + gg + 8 * lh;
        attnT[n * HID_ + m] = (bf16)acc1[nt][gg];
      }
    }
  }
  __syncthreads();

  // GEMM2: out[256, 64] = w_out(bf16) @ attn ; wave handles m-tiles {2wv,2wv+1}
  floatx8 acc2[2][4];
#pragma unroll
  for (int r = 0; r < 2; ++r)
#pragma unroll
    for (int nt = 0; nt < 4; ++nt) acc2[r][nt] = zero8();

#pragma unroll
  for (int kb_ = 0; kb_ < HID_ / 32; ++kb_) {
    bf16x16 Bfr[4];
#pragma unroll
    for (int nt = 0; nt < 4; ++nt)
      Bfr[nt] = *(const bf16x16*)&attnT[(nt * 16 + lr) * HID_ + kb_ * 32 + lh * 16];
    bf16x16 Afr[2];
#pragma unroll
    for (int r = 0; r < 2; ++r) {
      int m = (2 * wv + r) * 16 + lr;
      const bf16* p = woutb + (size_t)m * HID_ + kb_ * 32 + lh * 16;
      bf16x8 lo = *(const bf16x8*)p;
      bf16x8 hi = *(const bf16x8*)(p + 8);
      Afr[r] = join8(lo, hi);
    }
#pragma unroll
    for (int r = 0; r < 2; ++r)
#pragma unroll
      for (int nt = 0; nt < 4; ++nt)
        acc2[r][nt] = wmma_bf16(Afr[r], Bfr[nt], acc2[r][nt]);
  }

  // bias, then per-column sum of squares (LDS float atomics)
#pragma unroll
  for (int r = 0; r < 2; ++r)
#pragma unroll
    for (int nt = 0; nt < 4; ++nt) {
      float part = 0.f;
#pragma unroll
      for (int gg = 0; gg < 8; ++gg) {
        int m = (2 * wv + r) * 16 + gg + 8 * lh;
        float v = acc2[r][nt][gg] + bias[m];
        acc2[r][nt][gg] = v;
        part += v * v;
      }
      atomicAdd(&ssq[nt * 16 + lr], part);
    }
  __syncthreads();

  // RMS-norm: out = val / max(||col||, eps) * g[m] * sqrt(256)
#pragma unroll
  for (int r = 0; r < 2; ++r)
#pragma unroll
    for (int nt = 0; nt < 4; ++nt) {
      int n = nt * 16 + lr;
      float fac = 16.0f / fmaxf(sqrtf(ssq[n]), 1e-12f);
#pragma unroll
      for (int gg = 0; gg < 8; ++gg) {
        int m = (2 * wv + r) * 16 + gg + 8 * lh;
        out[((size_t)b * C_ + m) * N_ + n0 + n] = acc2[r][nt][gg] * gvec[m] * fac;
      }
    }
}

// ---------------------------------------------------------------------------
extern "C" void kernel_launch(void* const* d_in, const int* in_sizes, int n_in,
                              void* d_out, int out_size, void* d_ws, size_t ws_size,
                              hipStream_t stream) {
  (void)in_sizes; (void)n_in; (void)out_size; (void)ws_size;
  const float* x     = (const float*)d_in[0];
  const float* w_qkv = (const float*)d_in[1];
  const float* w_out = (const float*)d_in[2];
  const float* b_out = (const float*)d_in[3];
  const float* gvec  = (const float*)d_in[4];
  float* out = (float*)d_out;

  char* ws = (char*)d_ws;
  bf16*  qb    = (bf16*)(ws + QB_OFF);
  bf16*  kb    = (bf16*)(ws + KB_OFF);
  bf16*  vb    = (bf16*)(ws + VB_OFF);
  bf16*  wqkvb = (bf16*)(ws + WQKV_OFF);
  bf16*  woutb = (bf16*)(ws + WOUT_OFF);
  float* kmax  = (float*)(ws + KMAX_OFF);
  float* ksum  = (float*)(ws + KSUM_OFF);
  float* ctxg  = (float*)(ws + CTX_OFF);

  la_prep  <<<384, 256, 0, stream>>>(w_qkv, w_out, wqkvb, woutb, ctxg);
  la_qkv   <<<B_ * (N_ / NT_), 256, 0, stream>>>(x, wqkvb, qb, kb, vb);
  la_kstats<<<B_ * HID_, 256, 0, stream>>>(kb, kmax, ksum);
  la_ctx   <<<B_ * H_ * KSPLIT_, 256, 0, stream>>>(kb, vb, kmax, ctxg);
  la_out   <<<B_ * (N_ / NT_), 256, 0, stream>>>(qb, woutb, ctxg, ksum, b_out, gvec, out);
}
